// ODFADV3_34677565948855
// MI455X (gfx1250) — compile-verified, matrix operand order checked by reading, not run on hardware
//
#include <hip/hip_runtime.h>

// ---------------------------------------------------------------------------
// Fused SDF-style MLP for MI455X (gfx1250, wave32, WMMA bf16 16x16x32).
//   binput = [coords(6) | emb[idx](256)]  (padded to 288)
//   6x (linear + relu + layernorm), skip-concat at layer 3 (K=544 padded)
//   heads: intersections = LN-out(layer1) @ Wi + bi ; depths = LN-out(layer5) @ Wd + bd
// All GEMMs in bf16 with f32 accumulation via v_wmma_f32_16x16x32_bf16.
//
// v3: 64 points per workgroup (uses CDNA5's 320KB LDS; ~70.5KB used).
// Each wave32 covers all FOUR 16-row M tiles and a disjoint pair of N tiles:
// every packed-weight B fragment is loaded once per workgroup and reused for
// 4 WMMAs -> L2 weight traffic = (P/64) * 0.95MB ~= 3.9GB, compute-bound.
// ---------------------------------------------------------------------------

typedef __attribute__((ext_vector_type(16))) __bf16        v16bf;
typedef __attribute__((ext_vector_type(8)))  float         v8f;
typedef __attribute__((ext_vector_type(4)))  unsigned int  u32x4;

union FragU {
    u32x4 q[2];   // 32 bytes
    v16bf bf;     // 16 bf16
};

__device__ __forceinline__ unsigned short f2bf(float f) {
    unsigned u = __float_as_uint(f);
    unsigned r = u + 0x7FFFu + ((u >> 16) & 1u);   // round-to-nearest-even
    return (unsigned short)(r >> 16);
}
__device__ __forceinline__ float bf2f(unsigned short h) {
    return __uint_as_float(((unsigned)h) << 16);
}

// ---------------------------------------------------------------------------
// Weight pre-pack: f32 [K][256] row-major -> bf16 in per-lane WMMA B-fragment
// order. Block = (ktile, ntile): 32 lanes x 8 dwords; lane holds column
// n = ntile*16 + (lane&15), K packing identical to the 16-bit A-matrix table
// (lanes 0-15: K {2w | 16+2(w-4)}, lanes 16-31: +8).
// Layout (ktiles): L0: 9 (K=262 pad 288) | L1,L2: 8 | L3: 17 (binput|x = 544)
//                  L4,L5: 8.   Total 58 ktiles * 4096 dwords = 237568 dwords.
// ---------------------------------------------------------------------------
__device__ __forceinline__ int map_k(int l, int k) {
    if (l == 0) return (k < 262) ? k : -1;
    if (l == 3) return (k < 262) ? k : ((k < 288) ? -1 : k - 26);
    return k;   // square layers: k < 256 always
}

__global__ __launch_bounds__(256)
void pack_weights_kernel(const float* __restrict__ W0, const float* __restrict__ W1,
                         const float* __restrict__ W2, const float* __restrict__ W3,
                         const float* __restrict__ W4, const float* __restrict__ W5,
                         unsigned* __restrict__ wpack) {
    const int nk[6] = {9, 8, 8, 17, 8, 8};
    const float* Ws[6] = {W0, W1, W2, W3, W4, W5};

    int gid = blockIdx.x * blockDim.x + threadIdx.x;
    int rem = gid;
    int l = 0;
    const float* W = nullptr;
    for (l = 0; l < 6; ++l) {
        int cnt = nk[l] * 4096;
        if (rem < cnt) { W = Ws[l]; break; }
        rem -= cnt;
    }
    if (l >= 6) return;

    int block  = rem >> 8;        // ktile*16 + ntile
    int within = rem & 255;
    int lane   = within >> 3;
    int w      = within & 7;
    int ktile  = block >> 4;
    int ntile  = block & 15;

    int n   = ntile * 16 + (lane & 15);
    int khi = (lane >> 4) * 8;
    int kb  = ((w < 4) ? (2 * w) : (16 + 2 * (w - 4))) + khi;
    int k0  = ktile * 32 + kb;

    int s0 = map_k(l, k0);
    int s1 = map_k(l, k0 + 1);
    float v0 = (s0 >= 0) ? W[(size_t)s0 * 256 + n] : 0.0f;
    float v1 = (s1 >= 0) ? W[(size_t)s1 * 256 + n] : 0.0f;
    unsigned d = (unsigned)f2bf(v0) | ((unsigned)f2bf(v1) << 16);
    wpack[gid] = d;
}

// ---------------------------------------------------------------------------
// Main fused kernel: 64 points per workgroup; each of 8 waves does
// M=64 (4 tiles) x N=32 (ntiles 2w, 2w+1).
// ---------------------------------------------------------------------------
#define XK    544   // 288 (padded binput) + 256 (x)
#define MROWS 64

__global__ __launch_bounds__(256)
void mlp_fused_kernel(const float* __restrict__ coords,
                      const int*   __restrict__ indices,
                      const float* __restrict__ emb,
                      const unsigned* __restrict__ wpack,
                      const float* __restrict__ b0, const float* __restrict__ b1,
                      const float* __restrict__ b2, const float* __restrict__ b3,
                      const float* __restrict__ b4, const float* __restrict__ b5,
                      const float* __restrict__ Wi, const float* __restrict__ bi,
                      const float* __restrict__ Wd, const float* __restrict__ bd,
                      float* __restrict__ out, int P)
{
    __shared__ __align__(16) unsigned short xcat[MROWS][XK];  // bf16 acts (68KB)
    __shared__ float psum[MROWS][4];
    __shared__ float psq[MROWS][4];
    __shared__ float s_mean[MROWS];
    __shared__ float s_rstd[MROWS];

    const int tid  = threadIdx.x;
    const int lane = tid & 31;
    const int wave = tid >> 5;       // 0..7 -> ntiles {2w, 2w+1}
    const int p0   = blockIdx.x * MROWS;

    // ---- build padded binput: 4 threads per row, 72 elems each (288 total)
    {
        int row = tid >> 2, seg = tid & 3;
        int p = p0 + row;
        int idx = indices[p];
        const float* cr = coords + (size_t)p * 6;
        const float* er = emb + (size_t)idx * 256;
        int kbeg = seg * 72;
        #pragma unroll 4
        for (int k = kbeg; k < kbeg + 72; ++k) {
            float v = 0.0f;
            if (k < 6)        v = cr[k];
            else if (k < 262) v = er[k - 6];
            xcat[row][k] = f2bf(v);
        }
    }
    __syncthreads();

    const int   ksA[6]   = {0, 9, 9, 0, 9, 9};
    const int   nkA[6]   = {9, 8, 8, 17, 8, 8};
    const int   woffA[6] = {0, 36864, 69632, 102400, 172032, 204800};
    const float* biasA[6] = {b0, b1, b2, b3, b4, b5};

    const int mlo = lane & 15;          // A-fragment row within a 16-row tile
    const int khi = (lane >> 4) * 8;    // A/B K sub-offset per lane half

    #pragma unroll
    for (int l = 0; l < 6; ++l) {
        const unsigned* wp = wpack + woffA[l];
        const float* bias  = biasA[l];

        // ---- accumulators seeded with bias (C layout: N = lane&15)
        // acc[u][v]: ntile u (0,1) x mtile v (0..3)
        v8f acc[2][4];
        #pragma unroll
        for (int u = 0; u < 2; ++u) {
            float bv = bias[(2 * wave + u) * 16 + (lane & 15)];
            v8f a;
            #pragma unroll
            for (int j = 0; j < 8; ++j) a[j] = bv;
            #pragma unroll
            for (int v = 0; v < 4; ++v) acc[u][v] = a;
        }

        // ---- GEMM: K loop over ktiles of 32 (bf16)
        for (int kt = 0; kt < nkA[l]; ++kt) {
            // Four A fragments (row tiles 0..3), from LDS
            FragU af[4];
            #pragma unroll
            for (int v = 0; v < 4; ++v) {
                const u32x4* ap =
                    (const u32x4*)&xcat[v * 16 + mlo][(ksA[l] + kt) * 32 + khi];
                af[v].q[0] = ap[0];   // K = khi .. khi+7
                af[v].q[1] = ap[2];   // K = 16+khi .. 16+khi+7  (+32 bytes)
            }
            // Two B fragments (ntiles 2w, 2w+1), each reused for 4 M tiles
            #pragma unroll
            for (int u = 0; u < 2; ++u) {
                FragU bfr;
                const u32x4* bp = (const u32x4*)(wp +
                    ((size_t)(kt * 16 + (2 * wave + u)) * 256 + lane * 8));
                bfr.q[0] = bp[0];
                bfr.q[1] = bp[1];
                #pragma unroll
                for (int v = 0; v < 4; ++v) {
                    acc[u][v] = __builtin_amdgcn_wmma_f32_16x16x32_bf16(
                        false, af[v].bf, false, bfr.bf, (short)0,
                        acc[u][v], false, false);
                }
            }
        }
        __syncthreads();   // all A reads done before overwriting x region

        // ---- ReLU + store bf16 into x region of xcat
        #pragma unroll
        for (int u = 0; u < 2; ++u) {
            int n = (2 * wave + u) * 16 + (lane & 15);
            #pragma unroll
            for (int v = 0; v < 4; ++v) {
                #pragma unroll
                for (int j = 0; j < 8; ++j) {
                    int mm = v * 16 + j + 8 * (lane >> 4);
                    float val = acc[u][v][j];
                    val = (val > 0.0f) ? val : 0.0f;
                    xcat[mm][288 + n] = f2bf(val);
                }
            }
        }
        __syncthreads();

        // ---- LayerNorm: stats (population var), 4 threads per row
        {
            int row = tid >> 2, seg = tid & 3;
            float s = 0.0f, sq = 0.0f;
            #pragma unroll 8
            for (int i = 0; i < 64; ++i) {
                float v = bf2f(xcat[row][288 + seg * 64 + i]);
                s += v; sq += v * v;
            }
            psum[row][seg] = s; psq[row][seg] = sq;
        }
        __syncthreads();
        if (tid < MROWS) {
            float s = 0.0f, sq = 0.0f;
            #pragma unroll
            for (int i = 0; i < 4; ++i) { s += psum[tid][i]; sq += psq[tid][i]; }
            float mean = s * (1.0f / 256.0f);
            float var  = sq * (1.0f / 256.0f) - mean * mean;
            s_mean[tid] = mean;
            s_rstd[tid] = rsqrtf(var + 1e-5f);
        }
        __syncthreads();
        {
            int row = tid >> 2, seg = tid & 3;
            float mean = s_mean[row], rstd = s_rstd[row];
            #pragma unroll 8
            for (int i = 0; i < 64; ++i) {
                int c = 288 + seg * 64 + i;
                xcat[row][c] = f2bf((bf2f(xcat[row][c]) - mean) * rstd);
            }
        }
        __syncthreads();

        // ---- heads: intersections after layer 1, depths after layer 5
        if (l == 1 || l == 5) {
            const float* Wh = (l == 1) ? Wi : Wd;
            const float* bh = (l == 1) ? bi : bd;
            int row = tid >> 2, seg = tid & 3;
            float s = 0.0f;
            #pragma unroll 8
            for (int i = 0; i < 64; ++i) {
                int n = seg * 64 + i;
                s += bf2f(xcat[row][288 + n]) * Wh[n];
            }
            psum[row][seg] = s;
            __syncthreads();
            if (tid < MROWS) {
                float acc2 = 0.0f;
                #pragma unroll
                for (int i = 0; i < 4; ++i) acc2 += psum[tid][i];
                out[((l == 1) ? 0 : P) + p0 + tid] = acc2 + bh[0];
            }
            __syncthreads();
        }
    }
}

// ---------------------------------------------------------------------------
extern "C" void kernel_launch(void* const* d_in, const int* in_sizes, int n_in,
                              void* d_out, int out_size, void* d_ws, size_t ws_size,
                              hipStream_t stream) {
    const float* coords = (const float*)d_in[0];
    const int*   indices= (const int*)  d_in[1];
    const float* emb    = (const float*)d_in[2];
    const float* W0 = (const float*)d_in[3];  const float* b0 = (const float*)d_in[4];
    const float* W1 = (const float*)d_in[5];  const float* b1 = (const float*)d_in[6];
    const float* W2 = (const float*)d_in[7];  const float* b2 = (const float*)d_in[8];
    const float* W3 = (const float*)d_in[9];  const float* b3 = (const float*)d_in[10];
    const float* W4 = (const float*)d_in[11]; const float* b4 = (const float*)d_in[12];
    const float* W5 = (const float*)d_in[13]; const float* b5 = (const float*)d_in[14];
    const float* Wi = (const float*)d_in[15]; const float* bi = (const float*)d_in[16];
    const float* Wd = (const float*)d_in[17]; const float* bd = (const float*)d_in[18];

    float* out = (float*)d_out;
    unsigned* wpack = (unsigned*)d_ws;   // needs 237568*4 = 950,272 B of d_ws

    int P = in_sizes[0] / 6;             // B*N points (262,144)

    const int PACK_DWORDS = 237568;
    pack_weights_kernel<<<(PACK_DWORDS + 255) / 256, 256, 0, stream>>>(
        W0, W1, W2, W3, W4, W5, wpack);

    mlp_fused_kernel<<<P / MROWS, 256, 0, stream>>>(
        coords, indices, emb, wpack,
        b0, b1, b2, b3, b4, b5, Wi, bi, Wd, bd, out, P);
}
